// TriangleAttention_31645319037638
// MI455X (gfx1250) — compile-verified, matrix operand order checked by reading, not run on hardware
//
#include <hip/hip_runtime.h>

typedef _Float16 half_t;
typedef __attribute__((ext_vector_type(16))) _Float16 v16h;
typedef __attribute__((ext_vector_type(8)))  _Float16 v8h;
typedef __attribute__((ext_vector_type(8)))  float    v8f;
typedef __attribute__((ext_vector_type(4)))  float    v4f;

#define C_IN    128
#define C_HID   32
#define N_HEADS 4
#define SEQ     256
#define NPOS    (SEQ*SEQ)
#define INF_    1000000000.0f

// ---------------------------------------------------------------------------
// WMMA helper: D = A(16x32 f16) * B(32x16 f16, given as N x K) + C (f32)
// ---------------------------------------------------------------------------
__device__ __forceinline__ v8f wmma16x16x32(v16h a, v16h b, v8f c) {
  return __builtin_amdgcn_wmma_f32_16x16x32_f16(
      /*neg_a=*/false, a, /*neg_b=*/false, b,
      /*c_mod=*/(short)0, c, /*reuse_a=*/false, /*reuse_b=*/false);
}

// Fragment load for the 16-bit operand layout (ISA 7.12.2):
// lane l: row = l&15, half_sel = l>>4; element e<8 -> K = hs*8+e,
// e>=8 -> K = 16+hs*8+(e-8). Both segments are contiguous 16B loads.
__device__ __forceinline__ v16h load_frag(const half_t* __restrict__ base,
                                          int ld, int row, int k0, int hs) {
  const half_t* p = base + (size_t)row * ld + k0 + hs * 8;
  v8h lo = *(const v8h*)(p);
  v8h hi = *(const v8h*)(p + 16);
  v16h r;
#pragma unroll
  for (int e = 0; e < 8; ++e) { r[e] = lo[e]; r[e + 8] = hi[e]; }
  return r;
}

// ---------------------------------------------------------------------------
// Kernel 1: LayerNorm (f32) -> f16 XL, fused triangle-bias projection (f32)
// One wave per (i,j) position; lane owns 4 channels.
// ---------------------------------------------------------------------------
__global__ __launch_bounds__(256) void ln_tri_kernel(
    const float* __restrict__ x, const float* __restrict__ ln_w,
    const float* __restrict__ ln_b, const float* __restrict__ w_tri,
    half_t* __restrict__ XL, float* __restrict__ tri) {
  int lane = threadIdx.x & 31;
  int wid  = threadIdx.x >> 5;
  int p = blockIdx.x * 8 + wid;

  v4f v = *(const v4f*)(x + (size_t)p * C_IN + lane * 4);
  float s  = v[0] + v[1] + v[2] + v[3];
  float s2 = v[0]*v[0] + v[1]*v[1] + v[2]*v[2] + v[3]*v[3];
#pragma unroll
  for (int off = 16; off; off >>= 1) {
    s  += __shfl_xor(s,  off, 32);
    s2 += __shfl_xor(s2, off, 32);
  }
  float mu  = s * (1.0f / 128.0f);
  float var = s2 * (1.0f / 128.0f) - mu * mu;
  float rs  = rsqrtf(var + 1e-5f);

  half_t* outp = XL + (size_t)p * C_IN + lane * 4;
  float t0 = 0.f, t1 = 0.f, t2 = 0.f, t3 = 0.f;
#pragma unroll
  for (int u = 0; u < 4; ++u) {
    int c = lane * 4 + u;
    float n = (v[u] - mu) * rs * ln_w[c] + ln_b[c];
    outp[u] = (half_t)n;
    t0 += n * w_tri[c * 4 + 0];
    t1 += n * w_tri[c * 4 + 1];
    t2 += n * w_tri[c * 4 + 2];
    t3 += n * w_tri[c * 4 + 3];
  }
#pragma unroll
  for (int off = 16; off; off >>= 1) {
    t0 += __shfl_xor(t0, off, 32); t1 += __shfl_xor(t1, off, 32);
    t2 += __shfl_xor(t2, off, 32); t3 += __shfl_xor(t3, off, 32);
  }
  if (lane == 0) {
    float* tp = tri + (size_t)p * 4;
    tp[0] = t0; tp[1] = t1; tp[2] = t2; tp[3] = t3;
  }
}

// ---------------------------------------------------------------------------
// Kernel 2: weight convert + transpose: W[K=128][N=128] f32 -> Wt[N][K] f16
// ---------------------------------------------------------------------------
__global__ void wconv_kernel(const float* __restrict__ W,
                             half_t* __restrict__ Wt) {
  int n = blockIdx.x;   // 0..127
  int k = threadIdx.x;  // 0..127
  Wt[(size_t)n * 128 + k] = (half_t)W[(size_t)k * 128 + n];
}

// ---------------------------------------------------------------------------
// Kernel 3: WMMA GEMM  Y[M,128] = X[M,128](f16) @ Wt^T (Wt is [128 n][128 k])
// Block = 128 threads (4 waves); wave computes a 16x32 tile, 4 K-steps.
// MODE 0: f16 out                (K, V projections)
// MODE 1: f16 out, * 1/sqrt(32)  (Q projection)
// MODE 2: f16 out, sigmoid(v+bg) (gate)
// MODE 3: f32 out, v + bo        (final output projection)
// ---------------------------------------------------------------------------
template <int MODE>
__global__ __launch_bounds__(128) void gemm128_kernel(
    const half_t* __restrict__ X, const half_t* __restrict__ Wt,
    const float* __restrict__ bias, half_t* __restrict__ Yh,
    float* __restrict__ Yf) {
  int lane = threadIdx.x & 31;
  int wid  = threadIdx.x >> 5;
  int hs   = lane >> 4;
  int l16  = lane & 15;
  int row0 = blockIdx.x * 64 + wid * 16;
  int col0 = blockIdx.y * 32;

  v8f c0 = {}, c1 = {};
#pragma unroll
  for (int kc = 0; kc < 4; ++kc) {
    int k0 = kc * 32;
    v16h a  = load_frag(X + (size_t)row0 * 128, 128, l16, k0, hs);
    v16h b0 = load_frag(Wt + (size_t)col0 * 128, 128, l16, k0, hs);
    v16h b1 = load_frag(Wt + (size_t)(col0 + 16) * 128, 128, l16, k0, hs);
    c0 = wmma16x16x32(a, b0, c0);
    c1 = wmma16x16x32(a, b1, c1);
  }
#pragma unroll
  for (int r = 0; r < 8; ++r) {
    int m = r + 8 * hs;  // row within tile
    float vals[2] = { c0[r], c1[r] };
#pragma unroll
    for (int t = 0; t < 2; ++t) {
      int n = col0 + l16 + t * 16;
      float v = vals[t];
      if (MODE == 1) v *= 0.17677669529663689f;      // 1/sqrt(C_HID)
      if (MODE == 2) v = 1.0f / (1.0f + __expf(-(v + bias[n])));
      if (MODE == 3) v += bias[n];
      size_t idx = (size_t)(row0 + m) * 128 + n;
      if (MODE == 3) Yf[idx] = v;
      else           Yh[idx] = (half_t)v;
    }
  }
}

// ---------------------------------------------------------------------------
// Kernel 4: attention for one (i, h, 32-query tile). 2 waves / block.
// S = Q K^T + mask_bias + tri_bias ; softmax ; O = P V ; gate ; f16 out.
// K tile staged via async global->LDS b128 DMA (ASYNCcnt path);
// V tile staged with b128 loads + LDS transpose scatter.
// LDS: K 16KB + V^T 16KB + scores 32KB + probs 16KB = 80KB.
// ---------------------------------------------------------------------------
#define AW 2
__global__ __launch_bounds__(64) void attn_kernel(
    const half_t* __restrict__ Q, const half_t* __restrict__ Km,
    const half_t* __restrict__ V, const half_t* __restrict__ G,
    const float* __restrict__ tri, const float* __restrict__ mask,
    half_t* __restrict__ OG) {
  __shared__ __align__(16) half_t Klds[SEQ * C_HID];
  __shared__ __align__(16) half_t Vt[C_HID * SEQ];
  __shared__ __align__(16) float  S[AW][16 * SEQ];
  __shared__ __align__(16) half_t P[AW][16 * SEQ];

  int tid  = threadIdx.x;
  int lane = tid & 31;
  int wid  = tid >> 5;
  int hs   = lane >> 4;
  int l16  = lane & 15;
  int i = blockIdx.z;
  int h = blockIdx.y;
  int q0 = blockIdx.x * (AW * 16) + wid * 16;  // query tile base in [0,256)

  // ---- stage K tile: 1024 chunks of 8 halves, async DMA global->LDS ------
  // chunk c -> key = c>>2, d8 = (c&3)*8 ; contiguous 16B on both sides.
  for (int c = tid; c < (SEQ * C_HID) / 8; c += AW * 32) {
    int key = c >> 2, d8 = (c & 3) * 8;
    unsigned long long gaddr = (unsigned long long)(const void*)(
        Km + ((size_t)(i * SEQ + key)) * C_IN + h * C_HID + d8);
    unsigned laddr = (unsigned)(unsigned long long)(const void*)(
        Klds + key * C_HID + d8);
    asm volatile("global_load_async_to_lds_b128 %0, %1, off"
                 :: "v"(laddr), "v"(gaddr) : "memory");
  }
  // ---- stage V^T tile: b128 gather + 8-way LDS transpose scatter ---------
  for (int c = tid; c < (SEQ * C_HID) / 8; c += AW * 32) {
    int key = c >> 2, d8 = (c & 3) * 8;
    v8h vv = *(const v8h*)(V + ((size_t)(i * SEQ + key)) * C_IN
                             + h * C_HID + d8);
#pragma unroll
    for (int u = 0; u < 8; ++u) Vt[(d8 + u) * SEQ + key] = vv[u];
  }
  asm volatile("s_wait_asynccnt 0x0" ::: "memory");
  __syncthreads();

  // Q fragment: 16 queries x 32 (single K-step; Q pre-scaled by 1/sqrt(32))
  v16h qf = load_frag(Q + ((size_t)(i * SEQ + q0)) * C_IN + h * C_HID,
                      C_IN, l16, 0, hs);

  // scores: 16 key-tiles of 16
  for (int nt = 0; nt < SEQ / 16; ++nt) {
    v16h kf = load_frag(Klds + nt * 16 * C_HID, C_HID, l16, 0, hs);
    v8f c = {};
    c = wmma16x16x32(qf, kf, c);
#pragma unroll
    for (int r = 0; r < 8; ++r) {
      int m   = r + 8 * hs;       // query within tile
      int key = nt * 16 + l16;    // key
      int qg  = q0 + m;           // query in [0,256)
      float b = INF_ * (mask[(size_t)i * SEQ + key] - 1.0f)
              + tri[((size_t)qg * SEQ + key) * 4 + h];
      S[wid][m * SEQ + key] = c[r] + b;
    }
  }
  __syncthreads();

  // row softmax (wave owns its 16 rows; lane holds 8 elems/row)
  for (int row = 0; row < 16; ++row) {
    float* sr = &S[wid][row * SEQ];
    float vals[8];
    float mx = -3.4e38f;
#pragma unroll
    for (int j = 0; j < 8; ++j) { vals[j] = sr[lane + j * 32]; mx = fmaxf(mx, vals[j]); }
#pragma unroll
    for (int off = 16; off; off >>= 1) mx = fmaxf(mx, __shfl_xor(mx, off, 32));
    float sum = 0.f;
#pragma unroll
    for (int j = 0; j < 8; ++j) { vals[j] = __expf(vals[j] - mx); sum += vals[j]; }
#pragma unroll
    for (int off = 16; off; off >>= 1) sum += __shfl_xor(sum, off, 32);
    float inv = 1.0f / sum;
#pragma unroll
    for (int j = 0; j < 8; ++j)
      P[wid][row * SEQ + lane + j * 32] = (half_t)(vals[j] * inv);
  }
  __syncthreads();

  // O(16x32) = P(16x256) @ V(256x32): 8 K-steps, two N-halves
  v8f o0 = {}, o1 = {};
#pragma unroll
  for (int kc = 0; kc < 8; ++kc) {
    v16h a  = load_frag(P[wid], SEQ, l16, kc * 32, hs);
    v16h b0 = load_frag(Vt,            SEQ, l16, kc * 32, hs);
    v16h b1 = load_frag(Vt + 16 * SEQ, SEQ, l16, kc * 32, hs);
    o0 = wmma16x16x32(a, b0, o0);
    o1 = wmma16x16x32(a, b1, o1);
  }

  // gate + store f16
#pragma unroll
  for (int r = 0; r < 8; ++r) {
    int m  = r + 8 * hs;
    int qg = q0 + m;
    size_t base = ((size_t)(i * SEQ + qg)) * C_IN + h * C_HID;
    float g0 = (float)G[base + l16];
    float g1 = (float)G[base + 16 + l16];
    OG[base + l16]      = (half_t)(o0[r] * g0);
    OG[base + 16 + l16] = (half_t)(o1[r] * g1);
  }
}

// ---------------------------------------------------------------------------
extern "C" void kernel_launch(void* const* d_in, const int* in_sizes, int n_in,
                              void* d_out, int out_size, void* d_ws, size_t ws_size,
                              hipStream_t stream) {
  const float* x     = (const float*)d_in[0];
  const float* mask  = (const float*)d_in[1];
  const float* ln_w  = (const float*)d_in[2];
  const float* ln_b  = (const float*)d_in[3];
  const float* w_tri = (const float*)d_in[4];
  const float* wq    = (const float*)d_in[5];
  const float* wk    = (const float*)d_in[6];
  const float* wv    = (const float*)d_in[7];
  const float* wg    = (const float*)d_in[8];
  const float* bg    = (const float*)d_in[9];
  const float* wo    = (const float*)d_in[10];
  const float* bo    = (const float*)d_in[11];
  (void)in_sizes; (void)n_in; (void)out_size; (void)ws_size;

  char* ws = (char*)d_ws;
  size_t off = 0;
  auto wsalloc = [&](size_t bytes) -> char* {
    char* p = ws + off;
    off += (bytes + 255) & ~(size_t)255;
    return p;
  };
  half_t* XL   = (half_t*)wsalloc((size_t)NPOS * C_IN * 2);
  half_t* Qh   = (half_t*)wsalloc((size_t)NPOS * C_IN * 2);
  half_t* Kh   = (half_t*)wsalloc((size_t)NPOS * C_IN * 2);
  half_t* Vh   = (half_t*)wsalloc((size_t)NPOS * C_IN * 2);
  half_t* Gh   = (half_t*)wsalloc((size_t)NPOS * C_IN * 2);
  half_t* OGh  = (half_t*)wsalloc((size_t)NPOS * C_IN * 2);
  float*  trib = (float*)wsalloc((size_t)NPOS * 4 * 4);
  half_t* Wt   = (half_t*)wsalloc((size_t)5 * 128 * 128 * 2);

  // 1) layernorm + tri bias
  ln_tri_kernel<<<NPOS / 8, 256, 0, stream>>>(x, ln_w, ln_b, w_tri, XL, trib);

  // 2) weight convert+transpose (f32 [K][N] -> f16 [N][K])
  wconv_kernel<<<128, 128, 0, stream>>>(wq, Wt + 0 * 16384);
  wconv_kernel<<<128, 128, 0, stream>>>(wk, Wt + 1 * 16384);
  wconv_kernel<<<128, 128, 0, stream>>>(wv, Wt + 2 * 16384);
  wconv_kernel<<<128, 128, 0, stream>>>(wg, Wt + 3 * 16384);
  wconv_kernel<<<128, 128, 0, stream>>>(wo, Wt + 4 * 16384);

  // 3) projections (Q scaled by 1/sqrt(C_HID); G = sigmoid(.+bg))
  dim3 gg(NPOS / 64, 4);
  gemm128_kernel<1><<<gg, 128, 0, stream>>>(XL, Wt + 0 * 16384, nullptr, Qh, nullptr);
  gemm128_kernel<0><<<gg, 128, 0, stream>>>(XL, Wt + 1 * 16384, nullptr, Kh, nullptr);
  gemm128_kernel<0><<<gg, 128, 0, stream>>>(XL, Wt + 2 * 16384, nullptr, Vh, nullptr);
  gemm128_kernel<2><<<gg, 128, 0, stream>>>(XL, Wt + 3 * 16384, bg,      Gh, nullptr);

  // 4) attention + gate
  dim3 ga(SEQ / (AW * 16), N_HEADS, SEQ);  // (8, 4, 256)
  attn_kernel<<<ga, AW * 32, 0, stream>>>(Qh, Kh, Vh, Gh, trib, mask, OGh);

  // 5) output projection + bias -> f32 d_out
  gemm128_kernel<3><<<gg, 128, 0, stream>>>(OGh, Wt + 4 * 16384, bo, nullptr, (float*)d_out);
}
